// E8Attention_12300786336401
// MI455X (gfx1250) — matrix operand order repaired and checked by reading, not compiled, not run on hardware
//
#include <hip/hip_runtime.h>

// ---------- types (clang ext vectors only; HIP vector structs have ctors that break unions) ----------
typedef __attribute__((ext_vector_type(16))) __bf16  v16bf;
typedef __attribute__((ext_vector_type(8)))  float   v8f;
typedef __attribute__((ext_vector_type(4)))  unsigned int u32x4;
typedef __attribute__((ext_vector_type(2)))  unsigned int u32x2;
typedef __attribute__((ext_vector_type(4)))  float   f32x4;

union V16 { u32x4 q[2]; v16bf v; };
union VU8 { u32x4 q; unsigned short u[8]; };

// ---------- CDNA5 async global->LDS path (guarded; sync fallback) ----------
#if __has_builtin(__builtin_amdgcn_global_load_async_to_lds_b128)
#define HAVE_ASYNC_LDS 1
#else
#define HAVE_ASYNC_LDS 0
#endif

#if HAVE_ASYNC_LDS
// Builtin signature (from compiler diagnostic): param0 is
// 'int __vector(4) __device__ *' (AS1), param1 is the LDS side (AS3).
typedef int v4i_ __attribute__((vector_size(16)));
typedef __attribute__((address_space(1))) v4i_* gvec4p;
typedef __attribute__((address_space(3))) v4i_* lvec4p;
__device__ inline void async_g2l_b128(const void* g, void* l) {
  __builtin_amdgcn_global_load_async_to_lds_b128((gvec4p)g, (lvec4p)l, 0, 0);
}
__device__ inline void wait_async0() {
#if __has_builtin(__builtin_amdgcn_s_wait_asynccnt)
  __builtin_amdgcn_s_wait_asynccnt(0);
#else
  asm volatile("s_wait_asynccnt 0" ::: "memory");
#endif
}
#endif

__device__ inline unsigned short f2bf(float f) {
  unsigned int u = __float_as_uint(f);
  u += 0x7fffu + ((u >> 16) & 1u);          // round-to-nearest-even
  return (unsigned short)(u >> 16);
}
__device__ inline float bf2f(unsigned short h) {
  return __uint_as_float(((unsigned int)h) << 16);
}
__device__ inline v8f wmma_bf16(v16bf a, v16bf b, v8f c) {
  return __builtin_amdgcn_wmma_f32_16x16x32_bf16(false, a, false, b, (short)0, c, false, false);
}
__device__ inline v8f vzero8() {
  v8f z;
#pragma unroll
  for (int i = 0; i < 8; ++i) z[i] = 0.0f;
  return z;
}
// A-operand (16x32 bf16, MxK): lanes 0-15 hold K {0..7,16..23}, lanes 16-31 hold K {8..15,24..31}
__device__ inline v16bf load_a16(const unsigned short* base, int laneHalf) {
  V16 a;
  a.q[0] = *(const u32x4*)(base + laneHalf * 8);
  a.q[1] = *(const u32x4*)(base + 16 + laneHalf * 8);
  return a.v;
}
// B-operand (32x16 bf16, KxN): lane holds column n, lanes 0-15 hold K {0..7,16..23}, lanes 16-31 hold K {8..15,24..31}
__device__ inline v16bf load_b16(const unsigned short* base, int laneHalf) {
  V16 b;
  b.q[0] = *(const u32x4*)(base + laneHalf * 8);
  b.q[1] = *(const u32x4*)(base + 16 + laneHalf * 8);
  return b.v;
}

// ====================================================================
// Generic bf16 WMMA GEMM: C[M,N] = A[M,K] * B[K,N]
// EPI==0: A is fp32 (x), epilogue scatters bf16 into qkv [3][B][H][T][64]
// EPI==1: A is bf16 (attn out, async-staged), epilogue writes fp32 C
// Block tile 128x128, 256 threads = 8 waves, wave tile 32(M) x 64(N),
// K-step 64 (16 WMMAs per wave per barrier pair)
// ====================================================================
template <int EPI>
__global__ __launch_bounds__(256) void gemm_bf16_kernel(
    const void* __restrict__ Aptr, const float* __restrict__ Bmat,
    void* __restrict__ Cout, int N, int K) {
  __shared__ __align__(16) unsigned short As[128 * 72];  // [m][k], stride 72 (pad)
  __shared__ __align__(16) unsigned short Bs[128 * 72];  // [n][k] transposed, stride 72

  const int tid  = threadIdx.x;
  const int lane = tid & 31, wid = tid >> 5;
  const int l15 = lane & 15, lh = lane >> 4;
  const int wm = (wid & 3) * 32, wn = (wid >> 2) * 64;
  const int m0 = blockIdx.y * 128, n0 = blockIdx.x * 128;

  v8f acc[2][4];
#pragma unroll
  for (int i = 0; i < 2; ++i)
#pragma unroll
    for (int j = 0; j < 4; ++j) acc[i][j] = vzero8();

  for (int k0 = 0; k0 < K; k0 += 64) {
    // ---- stage A tile: 128 rows x 64 k ----
    if constexpr (EPI == 0) {
#pragma unroll
      for (int it = 0; it < 8; ++it) {
        int t = tid + it * 256;               // 0..2047
        int row = t >> 4, cg = (t & 15) * 4;  // 128 rows x 16 groups of 4 floats
        const float* src = (const float*)Aptr + (size_t)(m0 + row) * K + k0 + cg;
        f32x4 f = *(const f32x4*)src;
        u32x2 p;
        p.x = (unsigned)f2bf(f.x) | ((unsigned)f2bf(f.y) << 16);
        p.y = (unsigned)f2bf(f.z) | ((unsigned)f2bf(f.w) << 16);
        *(u32x2*)&As[row * 72 + cg] = p;
        if (k0 + 64 < K) __builtin_prefetch(src + 64, 0, 1);
      }
    } else {
#if HAVE_ASYNC_LDS
#pragma unroll
      for (int it = 0; it < 4; ++it) {
        int t = tid + it * 256;               // 0..1023
        int row = t >> 3, dg = (t & 7) * 8;   // 128 rows x 8 chunks of 8 bf16 (16B)
        async_g2l_b128((const unsigned short*)Aptr + (size_t)(m0 + row) * K + k0 + dg,
                       &As[row * 72 + dg]);
      }
#else
#pragma unroll
      for (int it = 0; it < 4; ++it) {
        int t = tid + it * 256;
        int row = t >> 3, dg = (t & 7) * 8;
        *(u32x4*)&As[row * 72 + dg] =
            *(const u32x4*)((const unsigned short*)Aptr + (size_t)(m0 + row) * K + k0 + dg);
      }
#endif
    }
    // ---- stage B tile transposed: 64 k-rows x 128 cols -> Bs[n][k] ----
#pragma unroll
    for (int it = 0; it < 8; ++it) {
      int t = tid + it * 256;                 // 0..2047
      int kr = t >> 5, ng = (t & 31) * 4;     // 64 k-rows x 32 groups of 4 floats
      const float* src = Bmat + (size_t)(k0 + kr) * N + n0 + ng;
      f32x4 f = *(const f32x4*)src;
      Bs[(ng + 0) * 72 + kr] = f2bf(f.x);
      Bs[(ng + 1) * 72 + kr] = f2bf(f.y);
      Bs[(ng + 2) * 72 + kr] = f2bf(f.z);
      Bs[(ng + 3) * 72 + kr] = f2bf(f.w);
      if (k0 + 64 < K) __builtin_prefetch(src + (size_t)64 * N, 0, 1);
    }
#if HAVE_ASYNC_LDS
    if constexpr (EPI == 1) wait_async0();
#endif
    __syncthreads();

    v16bf av[2][2];
#pragma unroll
    for (int i = 0; i < 2; ++i)
#pragma unroll
      for (int c = 0; c < 2; ++c)
        av[i][c] = load_a16(&As[(wm + i * 16 + l15) * 72 + c * 32], lh);
#pragma unroll
    for (int j = 0; j < 4; ++j) {
      v16bf b0 = load_b16(&Bs[(wn + j * 16 + l15) * 72], lh);
      v16bf b1 = load_b16(&Bs[(wn + j * 16 + l15) * 72 + 32], lh);
#pragma unroll
      for (int i = 0; i < 2; ++i) {
        acc[i][j] = wmma_bf16(av[i][0], b0, acc[i][j]);
        acc[i][j] = wmma_bf16(av[i][1], b1, acc[i][j]);
      }
    }
    __syncthreads();
  }

  // ---- epilogue: C layout row = r + 8*(lane>=16), col = lane&15 ----
#pragma unroll
  for (int i = 0; i < 2; ++i)
#pragma unroll
    for (int j = 0; j < 4; ++j)
#pragma unroll
      for (int r = 0; r < 8; ++r) {
        int gr = m0 + wm + i * 16 + r + lh * 8;
        int gc = n0 + wn + j * 16 + l15;
        float val = acc[i][j][r];
        if constexpr (EPI == 0) {
          // scatter into qkv bf16: [3][B=2][H=16][T=2048][HD=64]
          unsigned short* qkvb = (unsigned short*)Cout;
          int three = gc >> 10, hh = (gc >> 6) & 15, dd = gc & 63;
          int bb = gr >> 11, tt = gr & 2047;
          size_t off = ((((size_t)three * 2 + bb) * 16 + hh) * 2048 + tt) * 64 + dd;
          qkvb[off] = f2bf(val);
        } else {
          ((float*)Cout)[(size_t)gr * N + gc] = val;
        }
      }
}

// ====================================================================
// q_proj / k_proj: dot(q[...,0:8], head_dirs[h]) per (b,h,t)
// ====================================================================
__global__ __launch_bounds__(256) void proj_kernel(
    const unsigned short* __restrict__ qkvb, const float* __restrict__ dirs,
    float* __restrict__ qproj, float* __restrict__ kproj) {
  int idx = blockIdx.x * 256 + threadIdx.x;  // [B][H][T] flat, 65536 total
  int h = (idx >> 11) & 15;
  const unsigned short* q = qkvb + (size_t)idx * 64;
  const unsigned short* k = qkvb + (size_t)(65536 + idx) * 64;  // three==1 section
  float sq = 0.f, sk = 0.f;
#pragma unroll
  for (int d = 0; d < 8; ++d) {
    float w = dirs[h * 8 + d];
    sq += bf2f(q[d]) * w;
    sk += bf2f(k[d]) * w;
  }
  qproj[idx] = sq;
  kproj[idx] = sk;
}

// ====================================================================
// Flash attention with E8 rank-1 bias. One block = (b, h, 128 q rows).
// 8 waves x 16 q rows. Key blocks of 64. WMMA for QK^T and PV.
// K tile staged via async global->LDS; V transposed through VGPRs.
// ====================================================================
__global__ __launch_bounds__(256) void attn_kernel(
    const unsigned short* __restrict__ qkvb,
    const float* __restrict__ qproj, const float* __restrict__ kproj,
    const float* __restrict__ head_scales,
    unsigned short* __restrict__ attn_out) {
  __shared__ __align__(16) unsigned short Ks[64 * 72];      // [key][d]   pad 72
  __shared__ __align__(16) unsigned short Vt[64 * 72];      // [d][key]   pad 72
  __shared__ __align__(16) unsigned short Pl[8 * 16 * 72];  // per-wave P, [row][key]
  __shared__ float kpS[64];

  const int tid = threadIdx.x;
  const int lane = tid & 31, w = tid >> 5;
  const int l15 = lane & 15, lh = lane >> 4, hl8 = lh * 8;
  const int qb = blockIdx.x, h = blockIdx.y, b = blockIdx.z;

  const size_t bh = (size_t)(b * 16 + h);
  const unsigned short* Qg = qkvb + (((size_t)b * 16 + h) * 2048 + (size_t)qb * 128) * 64;
  const unsigned short* Kg = qkvb + (((size_t)(2 + b) * 16 + h) * 2048) * 64;
  const unsigned short* Vg = qkvb + (((size_t)(4 + b) * 16 + h) * 2048) * 64;
  const float* qpg = qproj + bh * 2048 + qb * 128;
  const float* kpg = kproj + bh * 2048;
  const float hs = head_scales[h];
  const float SCALE_F = 0.125f;  // 1/sqrt(64)

  // Q rows for this wave, in A-operand layout, two K chunks (d 0..31, 32..63)
  v16bf qa[2];
#pragma unroll
  for (int c = 0; c < 2; ++c)
    qa[c] = load_a16(Qg + (size_t)(w * 16 + l15) * 64 + c * 32, lh);

  float qpv[8], mst[8], lst[8];
#pragma unroll
  for (int r = 0; r < 8; ++r) {
    qpv[r] = qpg[w * 16 + r + hl8];
    mst[r] = -1e30f;
    lst[r] = 0.f;
  }
  v8f oacc[4];
#pragma unroll
  for (int od = 0; od < 4; ++od) oacc[od] = vzero8();

  const int nkb = 2 * qb + 2;  // causal: keys 0 .. qb*128+127
  for (int kb = 0; kb < nkb; ++kb) {
    __syncthreads();  // previous block's LDS reads done
    // ---- stage K [64 keys][64 d] (async path) ----
#if HAVE_ASYNC_LDS
#pragma unroll
    for (int it = 0; it < 2; ++it) {
      int t = tid + it * 256;
      int key = t >> 3, dg = (t & 7) * 8;
      async_g2l_b128(Kg + (size_t)(kb * 64 + key) * 64 + dg, &Ks[key * 72 + dg]);
    }
#else
#pragma unroll
    for (int it = 0; it < 2; ++it) {
      int t = tid + it * 256;
      int key = t >> 3, dg = (t & 7) * 8;
      *(u32x4*)&Ks[key * 72 + dg] =
          *(const u32x4*)(Kg + (size_t)(kb * 64 + key) * 64 + dg);
    }
#endif
    // ---- stage V transposed [64 d][64 keys] ----
#pragma unroll
    for (int it = 0; it < 2; ++it) {
      int t = tid + it * 256;
      int key = t >> 3, dg = (t & 7) * 8;
      VU8 vr;
      vr.q = *(const u32x4*)(Vg + (size_t)(kb * 64 + key) * 64 + dg);
#pragma unroll
      for (int j = 0; j < 8; ++j) Vt[(dg + j) * 72 + key] = vr.u[j];
    }
    if (tid < 64) kpS[tid] = kpg[kb * 64 + tid];
#if HAVE_ASYNC_LDS
    wait_async0();
#endif
    __syncthreads();

    // ---- scores: 4 tiles of 16x16, K-dim 64 (2 WMMAs each) ----
    v8f s[4];
#pragma unroll
    for (int t = 0; t < 4; ++t) {
      v8f a = vzero8();
#pragma unroll
      for (int c = 0; c < 2; ++c) {
        v16bf bo = load_b16(&Ks[(t * 16 + l15) * 72 + c * 32], lh);
        a = wmma_bf16(qa[c], bo, a);
      }
      s[t] = a;
    }
    // ---- scale + E8 rank-1 bias + causal mask ----
    const bool diag = (kb >= 2 * qb);
#pragma unroll
    for (int t = 0; t < 4; ++t) {
      float kpv = kpS[t * 16 + l15];
      int colg = kb * 64 + t * 16 + l15;
#pragma unroll
      for (int r = 0; r < 8; ++r) {
        float v = s[t][r] * SCALE_F + hs * qpv[r] * kpv;
        if (diag) {
          int rowg = qb * 128 + w * 16 + r + hl8;
          if (colg > rowg) v = -1e30f;
        }
        s[t][r] = v;
      }
    }
    // ---- online softmax (row spans 16 lanes of one half-wave) ----
#pragma unroll
    for (int r = 0; r < 8; ++r) {
      float lmax = s[0][r];
#pragma unroll
      for (int t = 1; t < 4; ++t) lmax = fmaxf(lmax, s[t][r]);
#pragma unroll
      for (int off = 1; off < 16; off <<= 1)
        lmax = fmaxf(lmax, __shfl_xor(lmax, off, 32));
      float mnew = fmaxf(mst[r], lmax);
      float alpha = __expf(mst[r] - mnew);
      float lsum = 0.f;
#pragma unroll
      for (int t = 0; t < 4; ++t) {
        float p = __expf(s[t][r] - mnew);
        s[t][r] = p;
        lsum += p;
      }
#pragma unroll
      for (int off = 1; off < 16; off <<= 1)
        lsum += __shfl_xor(lsum, off, 32);
      lst[r] = lst[r] * alpha + lsum;
      mst[r] = mnew;
#pragma unroll
      for (int od = 0; od < 4; ++od) oacc[od][r] *= alpha;
    }
    // ---- write P into wave-private LDS (C-layout -> A-layout relay) ----
    unsigned short* Pw = &Pl[w * 16 * 72];
#pragma unroll
    for (int t = 0; t < 4; ++t)
#pragma unroll
      for (int r = 0; r < 8; ++r)
        Pw[(r + hl8) * 72 + t * 16 + l15] = f2bf(s[t][r]);
    // ---- P(16x64) @ V(64x64): 4 out tiles x 2 K chunks ----
    v16bf ao[2];
#pragma unroll
    for (int kc = 0; kc < 2; ++kc) ao[kc] = load_a16(&Pw[l15 * 72 + kc * 32], lh);
#pragma unroll
    for (int od = 0; od < 4; ++od)
#pragma unroll
      for (int kc = 0; kc < 2; ++kc) {
        v16bf bo = load_b16(&Vt[(od * 16 + l15) * 72 + kc * 32], lh);
        oacc[od] = wmma_bf16(ao[kc], bo, oacc[od]);
      }
  }

  // ---- normalize + write bf16 to attn_out [B*T, C] at col h*64+d ----
#pragma unroll
  for (int od = 0; od < 4; ++od)
#pragma unroll
    for (int r = 0; r < 8; ++r) {
      int rq = w * 16 + r + hl8;
      float val = oacc[od][r] / lst[r];
      size_t row = (size_t)(b * 2048 + qb * 128 + rq);
      attn_out[row * 1024 + h * 64 + od * 16 + l15] = f2bf(val);
    }
}

// ====================================================================
extern "C" void kernel_launch(void* const* d_in, const int* in_sizes, int n_in,
                              void* d_out, int out_size, void* d_ws, size_t ws_size,
                              hipStream_t stream) {
  (void)in_sizes; (void)n_in; (void)out_size; (void)ws_size;
  const float* x           = (const float*)d_in[0];
  const float* w_qkv       = (const float*)d_in[1];
  const float* w_out       = (const float*)d_in[2];
  const float* head_scales = (const float*)d_in[3];
  const float* head_dirs   = (const float*)d_in[4];
  float* out = (float*)d_out;

  char* ws = (char*)d_ws;
  unsigned short* qkvb = (unsigned short*)ws;                // 3*2*16*2048*64 bf16 = 25165824 B
  unsigned short* attn = (unsigned short*)(ws + 25165824);   // 4096*1024 bf16    =  8388608 B
  float* qproj = (float*)(ws + 33554432);                    // 65536 f32
  float* kproj = (float*)(ws + 33816576);                    // 65536 f32

  // 1) qkv = x @ w_qkv  (M=4096, N=3072, K=1024), scatter to bf16 [3][B][H][T][64]
  gemm_bf16_kernel<0><<<dim3(3072 / 128, 4096 / 128), 256, 0, stream>>>(
      x, w_qkv, (void*)qkvb, 3072, 1024);
  // 2) E8 projections
  proj_kernel<<<65536 / 256, 256, 0, stream>>>(qkvb, head_dirs, qproj, kproj);
  // 3) flash attention with rank-1 geo bias
  attn_kernel<<<dim3(16, 16, 2), 256, 0, stream>>>(qkvb, qproj, kproj, head_scales, attn);
  // 4) out = attn @ w_out (M=4096, N=1024, K=1024), fp32 out
  gemm_bf16_kernel<1><<<dim3(1024 / 128, 4096 / 128), 256, 0, stream>>>(
      (const void*)attn, w_out, (void*)out, 1024, 1024);
}
